// VQVAEConvZeroDSUS_PaperVersion_8349416423745
// MI455X (gfx1250) — matrix operand order repaired and checked by reading, not compiled, not run on hardware
//
#include <hip/hip_runtime.h>
#include <math.h>

// ---------------- problem constants ----------------
#define B_   32
#define T_   2048
#define T2_  1024
#define DIM_ 300
#define C_   256
#define NE_  256
#define CINP_ 320        // enc_c0 input channels padded 300->320
#define COUTP_ 320       // dec_out output rows padded 300->320
#define TP1_ (T_ + 2)    // 2050, halo
#define TP2_ (T2_ + 2)   // 1026, halo
#define NROWS_ (B_ * T2_)  // 32768 quantizer rows

typedef __attribute__((ext_vector_type(16))) __bf16 v16bf;
typedef __attribute__((ext_vector_type(8)))  float  v8f;

union FragBF { uint4 q[2]; v16bf v; };

__device__ __forceinline__ unsigned short f2bf(float f) {
  unsigned int u = __float_as_uint(f);
  unsigned int r = ((u >> 16) & 1u) + 0x7FFFu;   // round-to-nearest-even
  return (unsigned short)((u + r) >> 16);
}
__device__ __forceinline__ float bf2f(unsigned short h) {
  return __uint_as_float(((unsigned int)h) << 16);
}

// ---------------- weight convert: fp32 [O][I][3] -> bf16 [Opad][3][Ipad] ----------------
__global__ void wcvt_kernel(const float* __restrict__ src, unsigned short* __restrict__ dst,
                            int O, int I, int Opad, int Ipad) {
  int total = Opad * 3 * Ipad;
  int i = blockIdx.x * 256 + threadIdx.x;
  if (i >= total) return;
  int ci = i % Ipad;
  int k  = (i / Ipad) % 3;
  int o  = i / (3 * Ipad);
  float v = 0.f;
  if (o < O && ci < I) v = src[((size_t)o * I + ci) * 3 + k];
  dst[i] = f2bf(v);
}

// ---------------- embed: fp32 [B][T][300] -> bf16 [B][2050][320] (halo+pad zero) ----------------
__global__ void embed_kernel(const float* __restrict__ in, unsigned short* __restrict__ xin) {
  size_t total = (size_t)B_ * TP1_ * CINP_;
  size_t i = (size_t)blockIdx.x * 256 + threadIdx.x;
  if (i >= total) return;
  int c = (int)(i % CINP_);
  int t = (int)((i / CINP_) % TP1_);
  int b = (int)(i / ((size_t)CINP_ * TP1_));
  float v = 0.f;
  if (t >= 1 && t <= T_ && c < DIM_)
    v = in[((size_t)b * T_ + (t - 1)) * DIM_ + c];
  xin[i] = f2bf(v);
}

// ---------------- zero halo rows (rows 0 and Tpad-1) of a bf16 [B][Tpad][Cp] buffer ---------
__global__ void halo_zero_kernel(unsigned short* __restrict__ buf, int Tpad, int Cp) {
  int total = B_ * 2 * Cp;
  int i = blockIdx.x * 256 + threadIdx.x;
  if (i >= total) return;
  int c = i % Cp;
  int r = (i / Cp) & 1;
  int b = i / (2 * Cp);
  int row = r ? (Tpad - 1) : 0;
  buf[((size_t)b * Tpad + row) * Cp + c] = 0;
}

__global__ void zero_f32_kernel(float* __restrict__ p, int n) {
  int i = blockIdx.x * 256 + threadIdx.x;
  if (i < n) p[i] = 0.f;
}

// ---------------- nearest-neighbor upsample x2: H[B][1026][256] -> F[B][2050][256] ----------
__global__ void upsample2_kernel(const unsigned short* __restrict__ src,
                                 unsigned short* __restrict__ dst) {
  size_t total = (size_t)B_ * T_ * C_;
  size_t i = (size_t)blockIdx.x * 256 + threadIdx.x;
  if (i >= total) return;
  int c = (int)(i % C_);
  int u = (int)((i / C_) % T_);
  int b = (int)(i / ((size_t)C_ * T_));
  dst[((size_t)b * TP1_ + (u + 1)) * C_ + c] =
      src[((size_t)b * TP2_ + ((u >> 1) + 1)) * C_ + c];
}

// ---------------- k=3 conv as 3 shifted GEMMs via WMMA bf16 ----------------
// Per wave: 16(co) x 64(t) fp32 tile -> 4x A-fragment reuse, 4 WMMA per inner step.
// x:[B][TinPad][CIN] bf16 halo, wgt:[Opad][3][CIN] bf16, dst:[B][Tout+2][Cout] bf16 halo
// fuse: 0=none(bias), 1=leaky(bias+y), 2=residual(bias+y+skip)
template <int CIN>
__global__ __launch_bounds__(256) void conv_k3_wmma(
    const unsigned short* __restrict__ x, const unsigned short* __restrict__ wgt,
    const float* __restrict__ bias, unsigned short* __restrict__ dst,
    const unsigned short* __restrict__ skip,
    int Cout, int TinPad, int Tout, int stride, int fuse) {
  extern __shared__ unsigned short lw[];   // [32][3][CIN]
  const int tid = threadIdx.x;
  const int bb = blockIdx.z;
  const int coBlk = blockIdx.y * 32;
  const int tBlk = blockIdx.x * 256;

  {  // stage weight slice to LDS
    const uint4* s = (const uint4*)(wgt + (size_t)coBlk * 3 * CIN);
    uint4* d = (uint4*)lw;
    const int n16 = (32 * 3 * CIN) >> 3;
    for (int i = tid; i < n16; i += 256) d[i] = s[i];
  }
  __syncthreads();

  const int w = tid >> 5;
  const int lane = tid & 31;
  const int half = lane >> 4;
  const int nIdx = lane & 15;
  const int coW = (w >> 2) * 16;         // 0 or 16
  const int t0 = tBlk + (w & 3) * 64;

  v8f acc[4];
  const v8f zero8 = {0.f,0.f,0.f,0.f,0.f,0.f,0.f,0.f};
#pragma unroll
  for (int j = 0; j < 4; ++j) acc[j] = zero8;

  const size_t xbase = (size_t)bb * TinPad * CIN;
  int tj[4];
#pragma unroll
  for (int j = 0; j < 4; ++j) tj[j] = t0 + 16 * j + nIdx;

  for (int k = 0; k < 3; ++k) {
    const unsigned short* xk[4];
#pragma unroll
    for (int j = 0; j < 4; ++j)
      xk[j] = x + xbase + (size_t)(tj[j] * stride + k) * CIN + 8 * half;
    const unsigned short* wk = lw + ((size_t)(coW + nIdx) * 3 + k) * CIN + 8 * half;
#pragma unroll
    for (int ci0 = 0; ci0 < CIN; ci0 += 32) {
      FragBF a;
      a.q[0] = *(const uint4*)(wk + ci0);       // A: K = {8h..+7, 16+8h..+7}
      a.q[1] = *(const uint4*)(wk + ci0 + 16);
      FragBF bf[4];
#pragma unroll
      for (int j = 0; j < 4; ++j) {             // B: K = {8h..+7, 16+8h..+7}
        bf[j].q[0] = *(const uint4*)(xk[j] + ci0);
        bf[j].q[1] = *(const uint4*)(xk[j] + ci0 + 16);
      }
#pragma unroll
      for (int j = 0; j < 4; ++j)
        acc[j] = __builtin_amdgcn_wmma_f32_16x16x32_bf16(false, a.v, false, bf[j].v,
                                                         (short)0, acc[j], false, false);
    }
  }

  const int coOut = coBlk + coW + 8 * half;   // 8 contiguous channels per lane
  float bv[8];
#pragma unroll
  for (int r = 0; r < 8; ++r) bv[r] = bias[coOut + r];

#pragma unroll
  for (int j = 0; j < 4; ++j) {
    int t = t0 + j * 16 + nIdx;
    size_t base = ((size_t)bb * (Tout + 2) + (t + 1)) * Cout + coOut;
    float v[8];
#pragma unroll
    for (int r = 0; r < 8; ++r) v[r] = acc[j][r] + bv[r];
    if (fuse == 1) {
#pragma unroll
      for (int r = 0; r < 8; ++r) v[r] = v[r] >= 0.f ? v[r] : 0.2f * v[r];
    } else if (fuse == 2) {
      uint4 sq = *(const uint4*)(skip + base);
      const unsigned short* sp = (const unsigned short*)&sq;
#pragma unroll
      for (int r = 0; r < 8; ++r) v[r] += bf2f(sp[r]);
    }
    unsigned short o[8];
#pragma unroll
    for (int r = 0; r < 8; ++r) o[r] = f2bf(v[r]);
    *(uint4*)(dst + base) = *(const uint4*)o;
  }
}

// ---------------- final conv (dec_out): bf16 in, fp32 out [B][T][300], guarded ----------------
__global__ __launch_bounds__(256) void conv_k3_wmma_out(
    const unsigned short* __restrict__ x, const unsigned short* __restrict__ wgt,
    const float* __restrict__ bias, float* __restrict__ dst,
    int CoutReal, int TinPad, int Tout) {
  constexpr int CIN = C_;
  extern __shared__ unsigned short lw[];
  const int tid = threadIdx.x;
  const int bb = blockIdx.z;
  const int coBlk = blockIdx.y * 32;
  const int tBlk = blockIdx.x * 256;
  {
    const uint4* s = (const uint4*)(wgt + (size_t)coBlk * 3 * CIN);
    uint4* d = (uint4*)lw;
    const int n16 = (32 * 3 * CIN) >> 3;
    for (int i = tid; i < n16; i += 256) d[i] = s[i];
  }
  __syncthreads();

  const int w = tid >> 5;
  const int lane = tid & 31;
  const int half = lane >> 4;
  const int nIdx = lane & 15;
  const int coW = (w >> 2) * 16;
  const int t0 = tBlk + (w & 3) * 64;

  v8f acc[4];
  const v8f zero8 = {0.f,0.f,0.f,0.f,0.f,0.f,0.f,0.f};
#pragma unroll
  for (int j = 0; j < 4; ++j) acc[j] = zero8;

  const size_t xbase = (size_t)bb * TinPad * CIN;
  int tj[4];
#pragma unroll
  for (int j = 0; j < 4; ++j) tj[j] = t0 + 16 * j + nIdx;

  for (int k = 0; k < 3; ++k) {
    const unsigned short* xk[4];
#pragma unroll
    for (int j = 0; j < 4; ++j)
      xk[j] = x + xbase + (size_t)(tj[j] + k) * CIN + 8 * half;
    const unsigned short* wk = lw + ((size_t)(coW + nIdx) * 3 + k) * CIN + 8 * half;
#pragma unroll
    for (int ci0 = 0; ci0 < CIN; ci0 += 32) {
      FragBF a;
      a.q[0] = *(const uint4*)(wk + ci0);
      a.q[1] = *(const uint4*)(wk + ci0 + 16);
      FragBF bf[4];
#pragma unroll
      for (int j = 0; j < 4; ++j) {
        bf[j].q[0] = *(const uint4*)(xk[j] + ci0);
        bf[j].q[1] = *(const uint4*)(xk[j] + ci0 + 16);
      }
#pragma unroll
      for (int j = 0; j < 4; ++j)
        acc[j] = __builtin_amdgcn_wmma_f32_16x16x32_bf16(false, a.v, false, bf[j].v,
                                                         (short)0, acc[j], false, false);
    }
  }

#pragma unroll
  for (int j = 0; j < 4; ++j) {
    int t = t0 + j * 16 + nIdx;
#pragma unroll
    for (int r = 0; r < 8; ++r) {
      int co = coBlk + coW + 8 * half + r;
      if (co < CoutReal)
        dst[((size_t)bb * Tout + t) * CoutReal + co] = acc[j][r] + bias[co];
    }
  }
}

// ---------------- VQ: one block per (b,t2) row ----------------
__global__ __launch_bounds__(256) void vq_assign_kernel(
    const unsigned short* __restrict__ z,   // [B][1026][256] bf16 halo
    const float* __restrict__ cb,           // [256][256] fp32
    unsigned short* __restrict__ zq,        // [B][1026][256] bf16 halo (decoder input)
    float* __restrict__ zq_out,             // [N][256] fp32 (z_q_st output)
    float* __restrict__ counts, float* __restrict__ lossAcc) {
  __shared__ float zsh[NE_];
  __shared__ float dsh[NE_];
  __shared__ int   ish[NE_];
  const int tid = threadIdx.x;
  const int n = blockIdx.x;
  const int b = n >> 10;
  const int t2 = n & (T2_ - 1);
  const size_t zoff = ((size_t)b * TP2_ + t2 + 1) * C_;
  zsh[tid] = bf2f(z[zoff + tid]);
  __syncthreads();

  float s = 0.f;  // ||c||^2 - 2 z.c  (argmin-equivalent distance)
  const float* cr = cb + (size_t)tid * C_;
#pragma unroll 8
  for (int c = 0; c < C_; ++c) {
    float wv = cr[c];
    s = fmaf(wv, wv - 2.f * zsh[c], s);
  }
  dsh[tid] = s;
  ish[tid] = tid;
  __syncthreads();
  for (int off = 128; off > 0; off >>= 1) {
    if (tid < off) {
      float d2 = dsh[tid + off];
      int i2 = ish[tid + off];
      if (d2 < dsh[tid] || (d2 == dsh[tid] && i2 < ish[tid])) {
        dsh[tid] = d2;
        ish[tid] = i2;
      }
    }
    __syncthreads();
  }
  const int idx = ish[0];
  __syncthreads();

  float q = cb[(size_t)idx * C_ + tid];
  zq_out[(size_t)n * C_ + tid] = q;            // z_q_st == z_q numerically
  zq[zoff + tid] = f2bf(q);
  float d = q - zsh[tid];
  dsh[tid] = d * d;
  __syncthreads();
  for (int off = 128; off > 0; off >>= 1) {
    if (tid < off) dsh[tid] += dsh[tid + off];
    __syncthreads();
  }
  if (tid == 0) {
    atomicAdd(lossAcc, dsh[0]);
    atomicAdd(&counts[idx], 1.0f);
  }
}

__global__ __launch_bounds__(256) void vq_final_kernel(
    const float* __restrict__ counts, const float* __restrict__ lossAcc,
    float* __restrict__ out_loss, float* __restrict__ out_perp) {
  __shared__ float sh[NE_];
  int tid = threadIdx.x;
  float e = counts[tid] / (float)NROWS_;
  sh[tid] = e * logf(e + 1e-10f);
  __syncthreads();
  for (int off = 128; off > 0; off >>= 1) {
    if (tid < off) sh[tid] += sh[tid + off];
    __syncthreads();
  }
  if (tid == 0) {
    *out_perp = expf(-sh[0]);
    *out_loss = (*lossAcc) * 2.0f / ((float)NROWS_ * (float)C_);  // (1+BETA)*MSE
  }
}

// ======================= host launch =======================
extern "C" void kernel_launch(void* const* d_in, const int* in_sizes, int n_in,
                              void* d_out, int out_size, void* d_ws, size_t ws_size,
                              hipStream_t stream) {
  (void)in_sizes; (void)n_in; (void)out_size; (void)ws_size;
  const float* inp = (const float*)d_in[0];
  const float* wsrc[13] = {(const float*)d_in[1],  (const float*)d_in[3],
                           (const float*)d_in[5],  (const float*)d_in[7],
                           (const float*)d_in[9],  (const float*)d_in[11],
                           (const float*)d_in[13], (const float*)d_in[15],
                           (const float*)d_in[17], (const float*)d_in[19],
                           (const float*)d_in[21], (const float*)d_in[23],
                           (const float*)d_in[25]};
  const float* bsrc[13] = {(const float*)d_in[2],  (const float*)d_in[4],
                           (const float*)d_in[6],  (const float*)d_in[8],
                           (const float*)d_in[10], (const float*)d_in[12],
                           (const float*)d_in[14], (const float*)d_in[16],
                           (const float*)d_in[18], (const float*)d_in[20],
                           (const float*)d_in[22], (const float*)d_in[24],
                           (const float*)d_in[26]};
  const float* codebook = (const float*)d_in[27];

  // ---- output regions ----
  float* out_rec = (float*)d_out;                                   // [B][T][300]
  const size_t REC = (size_t)B_ * T_ * DIM_;
  float* out_loss = out_rec + REC;
  float* out_perp = out_loss + 1;
  float* out_zq = out_perp + 1;                                     // [B][T2][256]

  // ---- workspace layout ----
  char* ws = (char*)d_ws;
  size_t off = 0;
  auto alloc = [&](size_t bytes) {
    size_t o = off;
    off = (off + bytes + 255) & ~(size_t)255;
    return o;
  };
  const size_t szF = (size_t)B_ * TP1_ * C_ * 2;     // bf16 full-T buffer
  const size_t szH = (size_t)B_ * TP2_ * C_ * 2;     // bf16 half-T buffer
  unsigned short* xin = (unsigned short*)(ws + alloc((size_t)B_ * TP1_ * CINP_ * 2));
  unsigned short* F0 = (unsigned short*)(ws + alloc(szF));
  unsigned short* F1 = (unsigned short*)(ws + alloc(szF));
  unsigned short* F2 = (unsigned short*)(ws + alloc(szF));
  unsigned short* F3 = (unsigned short*)(ws + alloc(szF));
  unsigned short* H0 = (unsigned short*)(ws + alloc(szH));
  unsigned short* H1 = (unsigned short*)(ws + alloc(szH));
  unsigned short* H2 = (unsigned short*)(ws + alloc(szH));
  unsigned short* H3 = (unsigned short*)(ws + alloc(szH));
  // weights: [0]=enc_c0 (256x3x320), [1..11] 256x3x256, [12]=dec_out (320x3x256)
  unsigned short* W[13];
  int Opad[13], Ipad[13], Oreal[13], Ireal[13];
  for (int i = 0; i < 13; ++i) { Opad[i] = C_; Ipad[i] = C_; Oreal[i] = C_; Ireal[i] = C_; }
  Ipad[0] = CINP_; Ireal[0] = DIM_;
  Opad[12] = COUTP_; Oreal[12] = DIM_;
  for (int i = 0; i < 13; ++i)
    W[i] = (unsigned short*)(ws + alloc((size_t)Opad[i] * 3 * Ipad[i] * 2));
  float* counts = (float*)(ws + alloc((NE_ + 1) * sizeof(float)));
  float* lossAcc = counts + NE_;

  // ---- 1) convert weights ----
  for (int i = 0; i < 13; ++i) {
    int total = Opad[i] * 3 * Ipad[i];
    wcvt_kernel<<<(total + 255) / 256, 256, 0, stream>>>(wsrc[i], W[i], Oreal[i],
                                                         Ireal[i], Opad[i], Ipad[i]);
  }
  // ---- 2) embed input + zero halos + zero accumulators ----
  {
    size_t total = (size_t)B_ * TP1_ * CINP_;
    embed_kernel<<<(unsigned)((total + 255) / 256), 256, 0, stream>>>(inp, xin);
  }
  unsigned short* fbufs[4] = {F0, F1, F2, F3};
  unsigned short* hbufs[4] = {H0, H1, H2, H3};
  for (int i = 0; i < 4; ++i) {
    halo_zero_kernel<<<(B_ * 2 * C_ + 255) / 256, 256, 0, stream>>>(fbufs[i], TP1_, C_);
    halo_zero_kernel<<<(B_ * 2 * C_ + 255) / 256, 256, 0, stream>>>(hbufs[i], TP2_, C_);
  }
  zero_f32_kernel<<<2, 256, 0, stream>>>(counts, NE_ + 1);

  // ---- conv launcher helper (per-wave 16x64 tile; block 32x256) ----
  auto conv = [&](const unsigned short* x, int widx, unsigned short* dst,
                  const unsigned short* skip, int Cin, int TinPad, int Tout,
                  int stride, int fuse) {
    dim3 grid(Tout / 256, Opad[widx] / 32, B_);
    size_t shmem = (size_t)32 * 3 * Cin * 2;
    if (Cin == CINP_) {
      conv_k3_wmma<CINP_><<<grid, 256, shmem, stream>>>(x, W[widx], bsrc[widx], dst,
                                                        skip, C_, TinPad, Tout, stride, fuse);
    } else {
      conv_k3_wmma<C_><<<grid, 256, shmem, stream>>>(x, W[widx], bsrc[widx], dst,
                                                     skip, C_, TinPad, Tout, stride, fuse);
    }
  };

  // ---- 3) encoder ----
  conv(xin, 0, F0, nullptr, CINP_, TP1_, T_, 1, 1);   // enc_c0 + leaky
  conv(F0, 1, F1, nullptr, C_, TP1_, T_, 1, 1);       // enc_r0 conv1 + leaky
  conv(F1, 2, F2, F0, C_, TP1_, T_, 1, 2);            // enc_r0 conv2 + residual
  conv(F2, 3, H0, nullptr, C_, TP1_, T2_, 2, 1);      // enc_c1 stride-2 + leaky
  conv(H0, 4, H1, nullptr, C_, TP2_, T2_, 1, 1);      // enc_r1 conv1 + leaky
  conv(H1, 5, H2, H0, C_, TP2_, T2_, 1, 2);           // enc_r1 conv2 + residual -> z

  // ---- 4) quantizer ----
  vq_assign_kernel<<<NROWS_, 256, 0, stream>>>(H2, codebook, H3, out_zq, counts, lossAcc);
  vq_final_kernel<<<1, 256, 0, stream>>>(counts, lossAcc, out_loss, out_perp);

  // ---- 5) decoder ----
  conv(H3, 6, H0, nullptr, C_, TP2_, T2_, 1, 1);      // dec_r0 conv1 + leaky
  conv(H0, 7, H1, H3, C_, TP2_, T2_, 1, 2);           // dec_r0 conv2 + residual
  {
    size_t total = (size_t)B_ * T_ * C_;
    upsample2_kernel<<<(unsigned)((total + 255) / 256), 256, 0, stream>>>(H1, F0);
  }
  conv(F0, 8, F1, nullptr, C_, TP1_, T_, 1, 0);       // dec_c0 (bias only)
  conv(F1, 9, F2, nullptr, C_, TP1_, T_, 1, 1);       // dec_r1 conv1 + leaky
  conv(F2, 10, F3, F1, C_, TP1_, T_, 1, 2);           // dec_r1 conv2 + residual
  conv(F3, 11, F0, nullptr, C_, TP1_, T_, 1, 1);      // dec_c1 + leaky
  {                                                    // dec_out -> fp32 output
    dim3 grid(T_ / 256, COUTP_ / 32, B_);
    size_t shmem = (size_t)32 * 3 * C_ * 2;
    conv_k3_wmma_out<<<grid, 256, shmem, stream>>>(F0, W[12], bsrc[12], out_rec,
                                                   DIM_, TP1_, T_);
  }
}
